// WhisperSelfAttention_68264210202862
// MI455X (gfx1250) — matrix operand order repaired and checked by reading, not compiled, not run on hardware
//
#include <hip/hip_runtime.h>

typedef __bf16 bf16_t;
typedef __attribute__((ext_vector_type(16))) __bf16 bf16x16;
typedef __attribute__((ext_vector_type(8)))  __bf16 bf16x8;
typedef __attribute__((ext_vector_type(8)))  float   f32x8;
typedef __attribute__((ext_vector_type(4)))  float   f32x4;

#define EMBED  1280
#define NHEAD  20
#define HDIM   64
#define BSZ    32
#define TLEN   448
#define MROWS  (BSZ * TLEN)      /* 14336 */
#define BHCNT  (BSZ * NHEAD)     /* 640   */
#define KVELEMS ((size_t)BHCNT * TLEN * HDIM)     /* 18,350,080 */
#define AWELEMS ((size_t)BHCNT * TLEN * TLEN)     /* 128,450,560 */

// ---------------------------------------------------------------------------
// WMMA helpers (CDNA5 gfx1250, wave32). Layouts per CDNA5 ISA.
// ---------------------------------------------------------------------------
static __device__ __forceinline__ f32x8 wmma_bf16(bf16x16 a, bf16x16 b, f32x8 c) {
  return __builtin_amdgcn_wmma_f32_16x16x32_bf16(false, a, false, b, (short)0, c,
                                                 false, false);
}

// A fragment: 16x32 bf16, row-major source. a_row = &A[row][0] for this lane's row.
static __device__ __forceinline__ bf16x16 frag_a(const bf16_t* __restrict__ a_row,
                                                 int kk, int hi) {
  const bf16_t* p = a_row + kk + hi * 16;
  bf16x8 x = *(const bf16x8*)(p);
  bf16x8 y = *(const bf16x8*)(p + 8);
  bf16x16 r;
#pragma unroll
  for (int i = 0; i < 8; ++i) { r[i] = x[i]; r[i + 8] = y[i]; }
  return r;
}

// B fragment: 32x16 bf16, B[k][n] = src[n][k]; lane n holds 16 contiguous K.
static __device__ __forceinline__ bf16x16 frag_b(const bf16_t* __restrict__ b_col,
                                                 int kk, int hi) {
  return *(const bf16x16*)(b_col + kk + hi * 16);
}

// A fragment sourced from fp32 (converted on load).
static __device__ __forceinline__ bf16x16 frag_a_f32(const float* __restrict__ a_row,
                                                     int kk, int hi) {
  const float* p = a_row + kk + hi * 16;
  f32x4 a0 = *(const f32x4*)(p);
  f32x4 a1 = *(const f32x4*)(p + 4);
  f32x4 a2 = *(const f32x4*)(p + 8);
  f32x4 a3 = *(const f32x4*)(p + 12);
  bf16x16 r;
#pragma unroll
  for (int i = 0; i < 4; ++i) {
    r[i]      = (bf16_t)a0[i];
    r[4 + i]  = (bf16_t)a1[i];
    r[8 + i]  = (bf16_t)a2[i];
    r[12 + i] = (bf16_t)a3[i];
  }
  return r;
}

// B fragment sourced from fp32 (converted on load).
static __device__ __forceinline__ bf16x16 frag_b_f32(const float* __restrict__ b_col,
                                                     int kk, int hi) {
  const float* p = b_col + kk + hi * 16;
  f32x4 a0 = *(const f32x4*)(p);
  f32x4 a1 = *(const f32x4*)(p + 4);
  f32x4 a2 = *(const f32x4*)(p + 8);
  f32x4 a3 = *(const f32x4*)(p + 12);
  bf16x16 r;
#pragma unroll
  for (int i = 0; i < 4; ++i) {
    r[i]      = (bf16_t)a0[i];
    r[4 + i]  = (bf16_t)a1[i];
    r[8 + i]  = (bf16_t)a2[i];
    r[12 + i] = (bf16_t)a3[i];
  }
  return r;
}

// ---------------------------------------------------------------------------
// CDNA5 async global->LDS copy (ASYNCcnt path), inline asm for portability.
// ---------------------------------------------------------------------------
static __device__ __forceinline__ void async_ld_b128(const bf16_t* g, bf16_t* l) {
  asm volatile("global_load_async_to_lds_b128 %0, %1, off"
               :
               : "v"((unsigned)(unsigned long long)l),
                 "v"((unsigned long long)g)
               : "memory");
}
#define ASYNC_WAIT(n) asm volatile("s_wait_asynccnt " #n ::: "memory")

// ---------------------------------------------------------------------------
// Stage a 64-col x 64-K bf16 B tile (8KB) into LDS with 256 threads, in
// FRAGMENT-LINEAR layout: piece(sub, nt, half, lane) -> 16B, so a wave's
// ds_load_b128 has consecutive lanes on consecutive 16B chunks (bank-conflict
// free, exactly one 256B sweep per half-wave).
//   global piece: col = nt*16 + (lane&15), K = kc + sub*32 + (lane>=16)*16 + half*8
// ---------------------------------------------------------------------------
static __device__ __forceinline__ void stage_b64(const bf16_t* __restrict__ gk,
                                                 int ldb, bf16_t* lbuf, int tid) {
#pragma unroll
  for (int j = 0; j < 2; ++j) {
    const int p = tid * 2 + j;              // 512 pieces x 16B
    const int sub = p >> 8;                 // which 32-K half of the chunk
    const int rem = p & 255;
    const int col = rem >> 2;               // 0..63
    const int kseg = rem & 3;               // 16B segment within the 64B row run
    const int nt = col >> 4, lo = col & 15;
    const int lane = ((kseg & 2) << 3) | lo;   // ksegs 2,3 belong to hi lanes
    const int half = kseg & 1;
    const int loff = sub * 4096 + ((nt * 2 + half) * 32 + lane) * 8;
    async_ld_b128(gk + (size_t)col * ldb + sub * 32 + kseg * 8, lbuf + loff);
  }
}

// Read one B fragment (nt) from the fragment-linear LDS tile (lbsub = buf + sub*4096).
static __device__ __forceinline__ bf16x16 frag_b_lds(const bf16_t* lbsub,
                                                     int nt, int lane) {
  bf16x8 x = *(const bf16x8*)(lbsub + (nt * 64 + lane) * 8);
  bf16x8 y = *(const bf16x8*)(lbsub + (nt * 64 + 32 + lane) * 8);
  bf16x16 r;
#pragma unroll
  for (int i = 0; i < 8; ++i) { r[i] = x[i]; r[i + 8] = y[i]; }
  return r;
}

// ---------------------------------------------------------------------------
// Kernel: fp32 -> bf16 conversion (vectorized x4)
// ---------------------------------------------------------------------------
__global__ __launch_bounds__(256) void wsa_cvt_bf16(const float* __restrict__ src,
                                                    bf16_t* __restrict__ dst, int n) {
  int i = (blockIdx.x * 256 + threadIdx.x) * 4;
  if (i + 3 < n) {
    f32x4 v = *(const f32x4*)(src + i);
#pragma unroll
    for (int j = 0; j < 4; ++j) dst[i + j] = (bf16_t)v[j];
  } else {
    for (int j = 0; j < 4 && i + j < n; ++j) dst[i + j] = (bf16_t)src[i + j];
  }
}

// ---------------------------------------------------------------------------
// Kernel: init V^T (bf16, [b,h,d,s]) from past_value fp32 [b,h,s,d]
// ---------------------------------------------------------------------------
__global__ __launch_bounds__(256) void wsa_init_vT(const float* __restrict__ pv,
                                                   bf16_t* __restrict__ vT) {
  size_t idx = (size_t)blockIdx.x * 256 + threadIdx.x;
  if (idx >= KVELEMS) return;
  int d = (int)(idx & 63);
  size_t rest = idx >> 6;
  int s = (int)(rest % TLEN);
  size_t bh = rest / TLEN;
  vT[(bh * HDIM + d) * TLEN + s] = (bf16_t)pv[idx];
}

// ---------------------------------------------------------------------------
// Kernel: fused QKV projection GEMM.
// Block = 256 threads (8 waves), block tile 256x64, wave tile 32x64.
// 64-K B chunks (8KB) double-buffered in LDS via async global->LDS copies;
// one barrier pair per 16 WMMAs per wave.
// grid.x = 56 (M blocks), grid.y = 60 (q:0..19, k:20..39, v:40..59)
// ---------------------------------------------------------------------------
__global__ __launch_bounds__(256) void wsa_qkv_gemm(
    const bf16_t* __restrict__ hs,                 // [14336,1280] bf16
    const bf16_t* __restrict__ wq,
    const bf16_t* __restrict__ wk,
    const bf16_t* __restrict__ wv,
    const float*  __restrict__ bq,
    const float*  __restrict__ bv,
    const int*    __restrict__ cache_pos,
    bf16_t* __restrict__ q_ws,                     // [b,h,t,d] bf16 (scaled)
    float*  __restrict__ key_out,                  // [b,h,s,d] fp32 (cache)
    float*  __restrict__ val_out,                  // [b,h,s,d] fp32 (cache)
    bf16_t* __restrict__ vT_ws) {                  // [b,h,d,s] bf16
  __shared__ __align__(32) bf16_t sB[2][8192];     // 2 x 16KB
  const int tid = threadIdx.x;
  const int wave = tid >> 5;
  const int lane = tid & 31;
  const int laneLo = lane & 15;
  const int hi = lane >> 4;
  const int m0 = blockIdx.x * 256 + wave * 32;
  const int wc = blockIdx.y;
  const int sel = wc / 20;                         // 0=q 1=k 2=v
  const int n0 = (wc % 20) * 64;
  const bf16_t* W = (sel == 0) ? wq : ((sel == 1) ? wk : wv);
  const bf16_t* a_row0 = hs + (size_t)(m0 + laneLo) * EMBED;
  const bf16_t* a_row1 = a_row0 + (size_t)16 * EMBED;
  const bf16_t* gB = W + (size_t)n0 * EMBED;       // + col*EMBED + k

  f32x8 acc0[4] = {}, acc1[4] = {};
  stage_b64(gB, EMBED, &sB[0][0], tid);            // prologue: chunk 0
  for (int kc = 0, it = 0; kc < EMBED; kc += 64, ++it) {
    if (kc + 64 < EMBED) {
      stage_b64(gB + kc + 64, EMBED, &sB[(it + 1) & 1][0], tid);
      ASYNC_WAIT(2);                               // current chunk landed
    } else {
      ASYNC_WAIT(0);
    }
    __syncthreads();
    __builtin_prefetch(a_row0 + kc + 256, 0, 0);
    const bf16_t* lb = &sB[it & 1][0];
#pragma unroll
    for (int sub = 0; sub < 2; ++sub) {
      const int kk = kc + sub * 32;
      bf16x16 a0 = frag_a(a_row0, kk, hi);
      bf16x16 a1 = frag_a(a_row1, kk, hi);
      const bf16_t* lbsub = lb + sub * 4096;
#pragma unroll
      for (int nt = 0; nt < 4; ++nt) {
        bf16x16 b = frag_b_lds(lbsub, nt, lane);
        acc0[nt] = wmma_bf16(a0, b, acc0[nt]);
        acc1[nt] = wmma_bf16(a1, b, acc1[nt]);
      }
    }
    __syncthreads();                               // buffer reuse fence
  }

  const int cp = *cache_pos;
#pragma unroll
  for (int g = 0; g < 2; ++g) {
#pragma unroll
    for (int nt = 0; nt < 4; ++nt) {
      const int col = n0 + nt * 16 + laneLo;
      const int h = col >> 6, d = col & 63;
#pragma unroll
      for (int r = 0; r < 8; ++r) {
        const int row = m0 + g * 16 + r + hi * 8;
        const int b = row / TLEN, t = row % TLEN;
        float v = (g == 0) ? acc0[nt][r] : acc1[nt][r];
        if (sel == 0) {
          v = (v + bq[col]) * 0.125f;              // HEAD_DIM^-0.5
          q_ws[(((size_t)(b * NHEAD + h)) * TLEN + t) * HDIM + d] = (bf16_t)v;
        } else if (sel == 1) {
          key_out[(((size_t)(b * NHEAD + h)) * TLEN + (cp + t)) * HDIM + d] = v;
        } else {
          v += bv[col];
          val_out[(((size_t)(b * NHEAD + h)) * TLEN + (cp + t)) * HDIM + d] = v;
          vT_ws[(((size_t)(b * NHEAD + h)) * HDIM + d) * TLEN + (cp + t)] = (bf16_t)v;
        }
      }
    }
  }
}

// ---------------------------------------------------------------------------
// Kernel: scores S = Q K^T + mask. One wave -> 16x64 tile of S.
// grid.x = 28*7 = 196 (mt*7+ng), grid.y = 640 (b*NHEAD+h)
// ---------------------------------------------------------------------------
__global__ __launch_bounds__(32) void wsa_scores(
    const bf16_t* __restrict__ q_ws,
    const float*  __restrict__ key_states,         // fp32 full cache [b,h,s,d]
    const float*  __restrict__ mask,               // [b,1,t,s]
    float*        __restrict__ attn_w) {
  const int lane = threadIdx.x & 31;
  const int laneLo = lane & 15;
  const int hi = lane >> 4;
  const int bh = blockIdx.y;
  const int b = bh / NHEAD;
  const int mt = blockIdx.x / 7, ng = blockIdx.x % 7;
  const int m0 = mt * 16, n0 = ng * 64;

  const bf16_t* a_row = q_ws + ((size_t)bh * TLEN + m0 + laneLo) * HDIM;
  const float* kbase = key_states + (size_t)bh * TLEN * HDIM;
  const float* b0 = kbase + (size_t)(n0 + 0 * 16 + laneLo) * HDIM;
  const float* b1 = kbase + (size_t)(n0 + 1 * 16 + laneLo) * HDIM;
  const float* b2 = kbase + (size_t)(n0 + 2 * 16 + laneLo) * HDIM;
  const float* b3 = kbase + (size_t)(n0 + 3 * 16 + laneLo) * HDIM;

  f32x8 acc[4] = {};
#pragma unroll
  for (int kk = 0; kk < HDIM; kk += 32) {
    bf16x16 a = frag_a(a_row, kk, hi);
    acc[0] = wmma_bf16(a, frag_b_f32(b0, kk, hi), acc[0]);
    acc[1] = wmma_bf16(a, frag_b_f32(b1, kk, hi), acc[1]);
    acc[2] = wmma_bf16(a, frag_b_f32(b2, kk, hi), acc[2]);
    acc[3] = wmma_bf16(a, frag_b_f32(b3, kk, hi), acc[3]);
  }
  const float* mrow = mask + (size_t)b * TLEN * TLEN;
  float* out = attn_w + (size_t)bh * TLEN * TLEN;
#pragma unroll
  for (int nt = 0; nt < 4; ++nt) {
#pragma unroll
    for (int r = 0; r < 8; ++r) {
      const int t = m0 + r + hi * 8;
      const int s = n0 + nt * 16 + laneLo;
      out[(size_t)t * TLEN + s] = acc[nt][r] + mrow[(size_t)t * TLEN + s];
    }
  }
}

// ---------------------------------------------------------------------------
// Kernel: row softmax over 448-wide rows. One wave per row, 14 elems/lane.
// ---------------------------------------------------------------------------
__global__ __launch_bounds__(256) void wsa_softmax(float* __restrict__ attn,
                                                   int nrows) {
  const int wave = (int)((blockIdx.x * 256 + threadIdx.x) >> 5);
  const int lane = threadIdx.x & 31;
  if (wave >= nrows) return;
  float* row = attn + (size_t)wave * TLEN;
  float vals[14];
  float m = -3.4e38f;
#pragma unroll
  for (int j = 0; j < 14; ++j) {
    vals[j] = row[lane + j * 32];
    m = fmaxf(m, vals[j]);
  }
#pragma unroll
  for (int off = 16; off > 0; off >>= 1) m = fmaxf(m, __shfl_xor(m, off, 32));
  float s = 0.f;
#pragma unroll
  for (int j = 0; j < 14; ++j) { vals[j] = __expf(vals[j] - m); s += vals[j]; }
#pragma unroll
  for (int off = 16; off > 0; off >>= 1) s += __shfl_xor(s, off, 32);
  const float inv = 1.0f / s;
#pragma unroll
  for (int j = 0; j < 14; ++j) row[lane + j * 32] = vals[j] * inv;
}

// ---------------------------------------------------------------------------
// Kernel: O = P @ V per (b,h). One wave -> 16x64 (full head dim) tile.
// grid.x = 28 (mt), grid.y = 640 (bh)
// ---------------------------------------------------------------------------
__global__ __launch_bounds__(32) void wsa_pv(
    const float*  __restrict__ attn_w,             // probs fp32 [b,h,t,s]
    const bf16_t* __restrict__ vT,                 // [b,h,d,s] bf16
    bf16_t*       __restrict__ o_ws) {             // [b,t,E] bf16
  const int lane = threadIdx.x & 31;
  const int laneLo = lane & 15;
  const int hi = lane >> 4;
  const int bh = blockIdx.y;
  const int b = bh / NHEAD, h = bh % NHEAD;
  const int m0 = blockIdx.x * 16;

  const float* a_row = attn_w + ((size_t)bh * TLEN + m0 + laneLo) * TLEN;
  const bf16_t* vbase = vT + (size_t)bh * HDIM * TLEN;
  const bf16_t* b0 = vbase + (size_t)(0 * 16 + laneLo) * TLEN;
  const bf16_t* b1 = vbase + (size_t)(1 * 16 + laneLo) * TLEN;
  const bf16_t* b2 = vbase + (size_t)(2 * 16 + laneLo) * TLEN;
  const bf16_t* b3 = vbase + (size_t)(3 * 16 + laneLo) * TLEN;

  f32x8 acc[4] = {};
  for (int kk = 0; kk < TLEN; kk += 32) {
    bf16x16 a = frag_a_f32(a_row, kk, hi);
    acc[0] = wmma_bf16(a, frag_b(b0, kk, hi), acc[0]);
    acc[1] = wmma_bf16(a, frag_b(b1, kk, hi), acc[1]);
    acc[2] = wmma_bf16(a, frag_b(b2, kk, hi), acc[2]);
    acc[3] = wmma_bf16(a, frag_b(b3, kk, hi), acc[3]);
  }
#pragma unroll
  for (int nt = 0; nt < 4; ++nt) {
#pragma unroll
    for (int r = 0; r < 8; ++r) {
      const int t = m0 + r + hi * 8;
      const int d = nt * 16 + laneLo;
      o_ws[((size_t)b * TLEN + t) * EMBED + h * HDIM + d] = (bf16_t)acc[nt][r];
    }
  }
}

// ---------------------------------------------------------------------------
// Kernel: output projection: attn_output = O @ Wo^T + bo (fp32 out).
// Same 8-wave / LDS-async structure as QKV. grid.x = 56, grid.y = 20.
// ---------------------------------------------------------------------------
__global__ __launch_bounds__(256) void wsa_oproj(
    const bf16_t* __restrict__ o_ws,               // [14336,1280] bf16
    const bf16_t* __restrict__ wo,                 // [1280,1280] bf16
    const float*  __restrict__ bo,
    float*        __restrict__ out) {              // [14336,1280] fp32
  __shared__ __align__(32) bf16_t sB[2][8192];
  const int tid = threadIdx.x;
  const int wave = tid >> 5;
  const int lane = tid & 31;
  const int laneLo = lane & 15;
  const int hi = lane >> 4;
  const int m0 = blockIdx.x * 256 + wave * 32;
  const int n0 = blockIdx.y * 64;
  const bf16_t* a_row0 = o_ws + (size_t)(m0 + laneLo) * EMBED;
  const bf16_t* a_row1 = a_row0 + (size_t)16 * EMBED;
  const bf16_t* gB = wo + (size_t)n0 * EMBED;

  f32x8 acc0[4] = {}, acc1[4] = {};
  stage_b64(gB, EMBED, &sB[0][0], tid);
  for (int kc = 0, it = 0; kc < EMBED; kc += 64, ++it) {
    if (kc + 64 < EMBED) {
      stage_b64(gB + kc + 64, EMBED, &sB[(it + 1) & 1][0], tid);
      ASYNC_WAIT(2);
    } else {
      ASYNC_WAIT(0);
    }
    __syncthreads();
    __builtin_prefetch(a_row0 + kc + 256, 0, 0);
    const bf16_t* lb = &sB[it & 1][0];
#pragma unroll
    for (int sub = 0; sub < 2; ++sub) {
      const int kk = kc + sub * 32;
      bf16x16 a0 = frag_a(a_row0, kk, hi);
      bf16x16 a1 = frag_a(a_row1, kk, hi);
      const bf16_t* lbsub = lb + sub * 4096;
#pragma unroll
      for (int nt = 0; nt < 4; ++nt) {
        bf16x16 b = frag_b_lds(lbsub, nt, lane);
        acc0[nt] = wmma_bf16(a0, b, acc0[nt]);
        acc1[nt] = wmma_bf16(a1, b, acc1[nt]);
      }
    }
    __syncthreads();
  }
#pragma unroll
  for (int g = 0; g < 2; ++g) {
#pragma unroll
    for (int nt = 0; nt < 4; ++nt) {
      const int col = n0 + nt * 16 + laneLo;
#pragma unroll
      for (int r = 0; r < 8; ++r) {
        const int row = m0 + g * 16 + r + hi * 8;
        const float v = (g == 0) ? acc0[nt][r] : acc1[nt][r];
        out[(size_t)row * EMBED + col] = v + bo[col];
      }
    }
  }
}

// ---------------------------------------------------------------------------
// Host launcher
// ---------------------------------------------------------------------------
extern "C" void kernel_launch(void* const* d_in, const int* in_sizes, int n_in,
                              void* d_out, int out_size, void* d_ws, size_t ws_size,
                              hipStream_t stream) {
  const float* hs     = (const float*)d_in[0];
  const float* past_k = (const float*)d_in[1];
  const float* past_v = (const float*)d_in[2];
  const float* mask   = (const float*)d_in[3];
  const float* Wq     = (const float*)d_in[4];
  const float* bq     = (const float*)d_in[5];
  const float* Wk     = (const float*)d_in[6];
  const float* Wv     = (const float*)d_in[7];
  const float* bv     = (const float*)d_in[8];
  const float* Wo     = (const float*)d_in[9];
  const float* bo     = (const float*)d_in[10];
  const int*   cpos   = (const int*)d_in[11];

  float* out       = (float*)d_out;
  float* attn_out  = out;                                  // 18,350,080
  float* attn_w    = out + (size_t)MROWS * EMBED;          // 128,450,560
  float* key_st    = attn_w + AWELEMS;                     // 18,350,080
  float* val_st    = key_st + KVELEMS;                     // 18,350,080

  // Workspace layout (all offsets multiples of 256 bytes)
  char* ws = (char*)d_ws;
  size_t off = 0;
  const size_t HS_BF = (size_t)MROWS * EMBED * sizeof(bf16_t);   // 36,700,160
  const size_t W_BF  = (size_t)EMBED * EMBED * sizeof(bf16_t);   //  3,276,800
  const size_t KV_BF = KVELEMS * sizeof(bf16_t);                 // 36,700,160
  bf16_t* hs_bf = (bf16_t*)(ws + off); off += HS_BF;
  bf16_t* wq_bf = (bf16_t*)(ws + off); off += W_BF;
  bf16_t* wk_bf = (bf16_t*)(ws + off); off += W_BF;
  bf16_t* wv_bf = (bf16_t*)(ws + off); off += W_BF;
  bf16_t* wo_bf = (bf16_t*)(ws + off); off += W_BF;
  bf16_t* q_bf  = (bf16_t*)(ws + off); off += KV_BF;
  bf16_t* vT_bf = (bf16_t*)(ws + off); off += KV_BF;
  bf16_t* o_bf  = (bf16_t*)(ws + off); off += HS_BF;
  (void)ws_size; (void)off; (void)in_sizes; (void)n_in; (void)out_size;

  // 1) Seed K/V caches with the past (graph-capture-safe async D2D copies).
  hipMemcpyAsync(key_st, past_k, KVELEMS * sizeof(float),
                 hipMemcpyDeviceToDevice, stream);
  hipMemcpyAsync(val_st, past_v, KVELEMS * sizeof(float),
                 hipMemcpyDeviceToDevice, stream);

  // 2) fp32 -> bf16 conversions.
  const int nHS = MROWS * EMBED;          // 18,350,080
  const int nW  = EMBED * EMBED;          //  1,638,400
  wsa_cvt_bf16<<<(nHS / 4 + 255) / 256, 256, 0, stream>>>(hs, hs_bf, nHS);
  wsa_cvt_bf16<<<(nW / 4 + 255) / 256, 256, 0, stream>>>(Wq, wq_bf, nW);
  wsa_cvt_bf16<<<(nW / 4 + 255) / 256, 256, 0, stream>>>(Wk, wk_bf, nW);
  wsa_cvt_bf16<<<(nW / 4 + 255) / 256, 256, 0, stream>>>(Wv, wv_bf, nW);
  wsa_cvt_bf16<<<(nW / 4 + 255) / 256, 256, 0, stream>>>(Wo, wo_bf, nW);
  wsa_init_vT<<<(unsigned)((KVELEMS + 255) / 256), 256, 0, stream>>>(past_v, vT_bf);

  // 3) Fused QKV projection (bf16 WMMA GEMM, M=14336, N=3*1280, K=1280).
  wsa_qkv_gemm<<<dim3(MROWS / 256, 60), 256, 0, stream>>>(
      hs_bf, wq_bf, wk_bf, wv_bf, bq, bv, cpos, q_bf, key_st, val_st, vT_bf);

  // 4) Attention scores + mask (per head GEMM, K from fp32 cache).
  wsa_scores<<<dim3((TLEN / 16) * (TLEN / 64), BHCNT), 32, 0, stream>>>(
      q_bf, key_st, mask, attn_w);

  // 5) Softmax over the 448-wide key axis.
  const int nrows = BHCNT * TLEN;         // 286,720
  wsa_softmax<<<(nrows * 32 + 255) / 256, 256, 0, stream>>>(attn_w, nrows);

  // 6) O = P @ V.
  wsa_pv<<<dim3(TLEN / 16, BHCNT), 32, 0, stream>>>(attn_w, vT_bf, o_bf);

  // 7) Output projection + bias.
  wsa_oproj<<<dim3(MROWS / 256, EMBED / 64), 256, 0, stream>>>(o_bf, wo_bf, bo,
                                                               attn_out);
}
